// MambaBlock_6055903888111
// MI455X (gfx1250) — matrix operand order repaired and checked by reading, not compiled, not run on hardware
//
#include <hip/hip_runtime.h>
#include <hip/hip_bf16.h>

// ---------------- problem constants ----------------
#define DM     768          // d_model
#define DI     1536         // d_inner
#define LSEQ   2048
#define BATCH  4
#define MROWS  (BATCH*LSEQ) // 8192 tokens
#define DTR    48           // dt_rank
#define NF     16           // fwd d_state
#define NR     4            // rev d_state
#define LDDBC  80           // row stride of DBC buffers (f32 and bf16)

typedef __attribute__((ext_vector_type(16))) __bf16 v16bf;
typedef __attribute__((ext_vector_type(8)))  __bf16 v8bf;
typedef __attribute__((ext_vector_type(8)))  float  v8f;

__device__ __forceinline__ __bf16 f2bf(float f) {
  union { float f; unsigned u; } x; x.f = f;
  unsigned u = x.u;
  unsigned r = u + 0x7FFFu + ((u >> 16) & 1u);   // round-to-nearest-even
  unsigned short s = (unsigned short)(r >> 16);
  union { unsigned short s; __bf16 b; } y; y.s = s;
  return y.b;
}

// =====================================================================
// bf16 WMMA GEMM:  C[8192,N] = epilogue( A[8192,Kpad] @ W[Npad,Kpad]^T )
// A, W already bf16; W zero-padded to Npad (mult of 128) x Kpad (mult 32),
// so the K-loop and staging have NO guards. Optional time-flip of A rows.
// Tiles are staged with GLOBAL_LOAD_ASYNC_TO_LDS_B128 (ASYNCcnt-tracked,
// no VGPR round trip), then consumed via ds_load_b128 fragment gathers.
// Block tile 128x128, 256 thr = 8 waves, wave = 32x64 C tile
//   -> 2x4 v_wmma_f32_16x16x32_bf16 accumulators (8 WMMA / K-tile / wave).
// epilogue: act 0 none, 1 bias+softplus; accum: C += ; Cb: optional bf16 copy.
// =====================================================================
__global__ __launch_bounds__(256)
void gemm_bf(const __bf16* __restrict__ A, int lda, int flipA,
             const __bf16* __restrict__ W, int ldw,
             const float* __restrict__ bias,
             float* C, __bf16* Cb, int ldc,
             int N, int kTiles, int accum, int act)
{
  constexpr int BM = 128, BN = 128, BK = 32, LS = 40;  // LS: padded LDS row (bf16)
  __shared__ __bf16 sA[BM * LS];
  __shared__ __bf16 sW[BN * LS];

  const int tid  = threadIdx.x;
  const int lane = tid & 31;
  const int wave = tid >> 5;
  const int wm   = (wave >> 1) * 32;    // 0,32,64,96
  const int wn   = (wave &  1) * 64;    // 0,64
  const int row  = lane & 15;
  const int half = lane >> 4;

  const int bM = blockIdx.y * BM;
  const int bN = blockIdx.x * BN;

  v8f acc[2][4] = {};

  for (int kt = 0; kt < kTiles; ++kt) {
    const int k0 = kt * BK;
    __syncthreads();   // previous iteration's fragment reads are done

    // ---- async stage A tile: 128x32 bf16, 2 x 16B per lane, global->LDS ----
    #pragma unroll
    for (int i = 0; i < 2; ++i) {
      int idx = tid + i * 256;          // 512 8-elem chunks
      int r = idx >> 2, c = idx & 3;
      long sr = bM + r;
      if (flipA) { int t = (int)(sr & (LSEQ - 1)); sr = sr - t + (LSEQ - 1 - t); }
      const __bf16* gp = &A[sr * (long)lda + k0 + c * 8];
      unsigned ldst = (unsigned)(size_t)&sA[r * LS + c * 8];
      asm volatile("global_load_async_to_lds_b128 %0, %1, off"
                   :: "v"(ldst), "v"(gp) : "memory");
    }
    // ---- async stage W tile: 128x32 bf16 (rows zero-padded in buffer) ----
    #pragma unroll
    for (int i = 0; i < 2; ++i) {
      int idx = tid + i * 256;
      int r = idx >> 2, c = idx & 3;
      const __bf16* gp = &W[(long)(bN + r) * ldw + k0 + c * 8];
      unsigned ldst = (unsigned)(size_t)&sW[r * LS + c * 8];
      asm volatile("global_load_async_to_lds_b128 %0, %1, off"
                   :: "v"(ldst), "v"(gp) : "memory");
    }
    asm volatile("s_wait_asynccnt 0x0" ::: "memory");
    __syncthreads();   // all waves' async transfers landed in LDS

    // ---- fragments per documented 16-bit WMMA VGPR layouts ----
    union FRAG { struct { v8bf lo, hi; } p; v16bf v; };
    v16bf afr[2], bfr[4];
    #pragma unroll
    for (int i = 0; i < 2; ++i) {       // A 16x32: K = (e>>3)*16 + half*8 + (e&7)
      const __bf16* pa = &sA[(wm + i * 16 + row) * LS + half * 8];
      FRAG t; t.p.lo = *(const v8bf*)pa; t.p.hi = *(const v8bf*)(pa + 16);
      afr[i] = t.v;
    }
    #pragma unroll
    for (int j = 0; j < 4; ++j) {       // B 32x16: col = lane&15, K = (e>>3)*16 + half*8 + (e&7)
      const __bf16* pb = &sW[(wn + j * 16 + row) * LS + half * 8];
      FRAG t; t.p.lo = *(const v8bf*)pb; t.p.hi = *(const v8bf*)(pb + 16);
      bfr[j] = t.v;
    }
    #pragma unroll
    for (int i = 0; i < 2; ++i)
      #pragma unroll
      for (int j = 0; j < 4; ++j)
        acc[i][j] = __builtin_amdgcn_wmma_f32_16x16x32_bf16(
            false, afr[i], false, bfr[j], (short)0, acc[i][j], false, false);
  }

  // ---- epilogue: C element (M = half*8 + r, N = lane&15) ----
  #pragma unroll
  for (int i = 0; i < 2; ++i) {
    #pragma unroll
    for (int j = 0; j < 4; ++j) {
      int gc = bN + wn + j * 16 + row;
      if (gc >= N) continue;
      float bval = bias ? bias[gc] : 0.f;
      #pragma unroll
      for (int r8 = 0; r8 < 8; ++r8) {
        int gr = bM + wm + i * 16 + half * 8 + r8;
        float v = acc[i][j][r8];
        if (act == 1) { v += bval; v = (v > 20.f) ? v : log1pf(__expf(v)); }
        long off = (long)gr * ldc + gc;
        if (accum) C[off] += v; else C[off] = v;
        if (Cb) Cb[off] = f2bf(v);
      }
    }
  }
}

// ---------------- f32 -> bf16 bulk convert ----------------
__global__ __launch_bounds__(256)
void cvt_x(const float* __restrict__ src, __bf16* __restrict__ dst, long n)
{
  long i = (long)blockIdx.x * 256 + threadIdx.x;
  if (i < n) dst[i] = f2bf(src[i]);
}

// ------- weight convert + zero-pad to [Npad x Kpad] bf16 -------
__global__ __launch_bounds__(256)
void cvt_w(const float* __restrict__ W, int N, int K,
           __bf16* __restrict__ Wb, int Npad, int Kpad)
{
  long i = (long)blockIdx.x * 256 + threadIdx.x;
  if (i >= (long)Npad * Kpad) return;
  int k = (int)(i % Kpad);
  int n = (int)(i / Kpad);
  float v = (n < N && k < K) ? W[(long)n * K + k] : 0.f;
  Wb[i] = f2bf(v);
}

// ---------------- causal depthwise conv(4) + SiLU, dual f32/bf16 out ----------------
__global__ __launch_bounds__(256)
void conv_silu(const float* __restrict__ XZ, const float* __restrict__ cw,
               const float* __restrict__ cb,
               float* __restrict__ XCf, __bf16* __restrict__ XCb)
{
  long i = (long)blockIdx.x * 256 + threadIdx.x;
  if (i >= (long)MROWS * DI) return;
  int  d = (int)(i % DI);
  long m = i / DI;
  int  t = (int)(m & (LSEQ - 1));
  float acc = cb[d];
  #pragma unroll
  for (int k = 0; k < 4; ++k) {
    int tt = t - 3 + k;
    if (tt >= 0) acc += XZ[(m - t + tt) * (long)(2 * DI) + d] * cw[d * 4 + k];
  }
  float s = acc / (1.f + __expf(-acc));   // SiLU
  XCf[i] = s;
  XCb[i] = f2bf(s);
}

// =====================================================================
// Selective scan + fused gate. One thread per (b,d); NS states in regs.
// B/C tiles (64 steps) staged in LDS. Emits bf16 gated output directly.
// =====================================================================
template <int NS>
__global__ __launch_bounds__(256)
void scan_gate(const float* __restrict__ DT, const float* __restrict__ XCf,
               const float* __restrict__ DBCf, const float* __restrict__ XZ,
               const float* __restrict__ A_log, const float* __restrict__ Dp,
               __bf16* __restrict__ Yb)
{
  constexpr int TCH = 64;
  __shared__ float sBC[TCH * 2 * NS];

  int gid = blockIdx.x * 256 + threadIdx.x;   // DI % 256 == 0 -> single b per block
  int b = gid / DI;
  int d = gid % DI;

  float Ar[NS], h[NS];
  #pragma unroll
  for (int n = 0; n < NS; ++n) { Ar[n] = -__expf(A_log[d * NS + n]); h[n] = 0.f; }
  const float Dval = Dp[d];

  const long rowBase = (long)b * LSEQ;
  for (int t0 = 0; t0 < LSEQ; t0 += TCH) {
    __syncthreads();
    for (int idx = threadIdx.x; idx < TCH * 2 * NS; idx += 256) {
      int tt = idx / (2 * NS);
      int c  = idx % (2 * NS);
      sBC[idx] = DBCf[(rowBase + t0 + tt) * (long)LDDBC + DTR + c];
    }
    __syncthreads();
    for (int tt = 0; tt < TCH; ++tt) {
      long r    = rowBase + t0 + tt;
      float dtv = DT[r * DI + d];
      float xv  = XCf[r * DI + d];
      const float* bc = &sBC[tt * 2 * NS];
      float y = 0.f;
      #pragma unroll
      for (int n = 0; n < NS; ++n) {
        float dA = __expf(dtv * Ar[n]);
        h[n] = dA * h[n] + (dtv * bc[n]) * xv;
        y   += h[n] * bc[NS + n];
      }
      float z = XZ[r * (long)(2 * DI) + DI + d];
      float g = (y + xv * Dval) * (z / (1.f + __expf(-z)));
      Yb[r * DI + d] = f2bf(g);
    }
  }
}

// =====================================================================
// Host-side orchestration
// =====================================================================
extern "C" void kernel_launch(void* const* d_in, const int* in_sizes, int n_in,
                              void* d_out, int out_size, void* d_ws, size_t ws_size,
                              hipStream_t stream) {
  const float* x      = (const float*)d_in[0];
  const float* inW_f  = (const float*)d_in[1];
  const float* cw_f   = (const float*)d_in[2];
  const float* cb_f   = (const float*)d_in[3];
  const float* xpW_f  = (const float*)d_in[4];
  const float* dtW_f  = (const float*)d_in[5];
  const float* dtb_f  = (const float*)d_in[6];
  const float* Alog_f = (const float*)d_in[7];
  const float* Dv_f   = (const float*)d_in[8];
  const float* outW_f = (const float*)d_in[9];
  const float* inW_r  = (const float*)d_in[10];
  const float* cw_r   = (const float*)d_in[11];
  const float* cb_r   = (const float*)d_in[12];
  const float* xpW_r  = (const float*)d_in[13];
  const float* dtW_r  = (const float*)d_in[14];
  const float* dtb_r  = (const float*)d_in[15];
  const float* Alog_r = (const float*)d_in[16];
  const float* Dv_r   = (const float*)d_in[17];
  const float* outW_r = (const float*)d_in[18];

  // -------- workspace carve-out (256B aligned) --------
  char* wp = (char*)d_ws;
  auto carve = [&](size_t bytes) -> char* {
    char* q = wp; wp += (bytes + 255) & ~(size_t)255; return q;
  };
  float*  XZ    = (float*) carve((size_t)MROWS * 2 * DI * 4);  // 100.7 MB
  float*  XCf   = (float*) carve((size_t)MROWS * DI * 4);      //  50.3 MB
  float*  DT    = (float*) carve((size_t)MROWS * DI * 4);      //  50.3 MB
  float*  DBCf  = (float*) carve((size_t)MROWS * LDDBC * 4);   //   2.6 MB
  __bf16* Xbf   = (__bf16*)carve((size_t)MROWS * DM * 2);      //  12.6 MB
  __bf16* XCb   = (__bf16*)carve((size_t)MROWS * DI * 2);      //  25.2 MB
  __bf16* DBCb  = (__bf16*)carve((size_t)MROWS * LDDBC * 2);   //   1.3 MB
  __bf16* Y0b   = (__bf16*)carve((size_t)MROWS * DI * 2);      //  25.2 MB
  __bf16* Y1b   = (__bf16*)carve((size_t)MROWS * DI * 2);      //  25.2 MB
  __bf16* inWb_f  = (__bf16*)carve((size_t)3072 * 768 * 2);
  __bf16* xpWb_f  = (__bf16*)carve((size_t)128 * 1536 * 2);
  __bf16* dtWb_f  = (__bf16*)carve((size_t)1536 * 64 * 2);
  __bf16* outWb_f = (__bf16*)carve((size_t)768 * 1536 * 2);
  __bf16* inWb_r  = (__bf16*)carve((size_t)3072 * 768 * 2);
  __bf16* xpWb_r  = (__bf16*)carve((size_t)128 * 1536 * 2);
  __bf16* dtWb_r  = (__bf16*)carve((size_t)1536 * 64 * 2);
  __bf16* outWb_r = (__bf16*)carve((size_t)768 * 1536 * 2);
  float* out = (float*)d_out;

  const dim3 blk(256);
  auto g1 = [](long n) { return dim3((unsigned)((n + 255) / 256)); };
  auto gg = [](int Npad) { return dim3(Npad / 128, MROWS / 128); };
  const long elems = (long)MROWS * DI;

  // -------- one-time converts (stateless, every call) --------
  cvt_x<<<g1((long)MROWS * DM), blk, 0, stream>>>(x, Xbf, (long)MROWS * DM);
  cvt_w<<<g1((long)3072 * 768), blk, 0, stream>>>(inW_f, 3072, 768, inWb_f, 3072, 768);
  cvt_w<<<g1((long)128 * 1536), blk, 0, stream>>>(xpW_f, DTR + 2 * NF, 1536, xpWb_f, 128, 1536);
  cvt_w<<<g1((long)1536 * 64),  blk, 0, stream>>>(dtW_f, 1536, DTR, dtWb_f, 1536, 64);
  cvt_w<<<g1((long)768 * 1536), blk, 0, stream>>>(outW_f, 768, 1536, outWb_f, 768, 1536);
  cvt_w<<<g1((long)3072 * 768), blk, 0, stream>>>(inW_r, 3072, 768, inWb_r, 3072, 768);
  cvt_w<<<g1((long)128 * 1536), blk, 0, stream>>>(xpW_r, DTR + 2 * NR, 1536, xpWb_r, 128, 1536);
  cvt_w<<<g1((long)1536 * 64),  blk, 0, stream>>>(dtW_r, 1536, DTR, dtWb_r, 1536, 64);
  cvt_w<<<g1((long)768 * 1536), blk, 0, stream>>>(outW_r, 768, 1536, outWb_r, 768, 1536);

  // ---------------- forward direction (N_state=16) ----------------
  gemm_bf<<<gg(3072), blk, 0, stream>>>(Xbf, DM, 0, inWb_f, DM, nullptr,
                                        XZ, nullptr, 2 * DI, 2 * DI, 768 / 32, 0, 0);
  conv_silu<<<g1(elems), blk, 0, stream>>>(XZ, cw_f, cb_f, XCf, XCb);
  gemm_bf<<<gg(128), blk, 0, stream>>>(XCb, DI, 0, xpWb_f, DI, nullptr,
                                       DBCf, DBCb, LDDBC, DTR + 2 * NF, 1536 / 32, 0, 0);
  gemm_bf<<<gg(1536), blk, 0, stream>>>(DBCb, LDDBC, 0, dtWb_f, 64, dtb_f,
                                        DT, nullptr, DI, DI, 64 / 32, 0, 1);
  scan_gate<NF><<<(BATCH * DI) / 256, blk, 0, stream>>>(DT, XCf, DBCf, XZ, Alog_f, Dv_f, Y0b);

  // ---------------- reverse direction (N_state=4), flip via GEMM ----------------
  gemm_bf<<<gg(3072), blk, 0, stream>>>(Xbf, DM, 1, inWb_r, DM, nullptr,
                                        XZ, nullptr, 2 * DI, 2 * DI, 768 / 32, 0, 0);
  conv_silu<<<g1(elems), blk, 0, stream>>>(XZ, cw_r, cb_r, XCf, XCb);
  gemm_bf<<<gg(128), blk, 0, stream>>>(XCb, DI, 0, xpWb_r, DI, nullptr,
                                       DBCf, DBCb, LDDBC, DTR + 2 * NR, 1536 / 32, 0, 0);
  gemm_bf<<<gg(1536), blk, 0, stream>>>(DBCb, LDDBC, 0, dtWb_r, 64, dtb_r,
                                        DT, nullptr, DI, DI, 64 / 32, 0, 1);
  scan_gate<NR><<<(BATCH * DI) / 256, blk, 0, stream>>>(DT, XCf, DBCf, XZ, Alog_r, Dv_r, Y1b);

  // ---------------- output projections: out = Y0@Wo_f^T + flip(Y1)@Wo_r^T ----------------
  gemm_bf<<<gg(768), blk, 0, stream>>>(Y0b, DI, 0, outWb_f, DI, nullptr,
                                       out, nullptr, DM, DM, 1536 / 32, 0, 0);
  gemm_bf<<<gg(768), blk, 0, stream>>>(Y1b, DI, 1, outWb_r, DI, nullptr,
                                       out, nullptr, DM, DM, 1536 / 32, 1, 0);
}
